// CCNN_29583734735381
// MI455X (gfx1250) — matrix operand order repaired and checked by reading, not compiled, not run on hardware
//
#include <hip/hip_runtime.h>
#include <hip/hip_bf16.h>

typedef __attribute__((ext_vector_type(16))) _Float16 v16h;
typedef __attribute__((ext_vector_type(8)))  float    v8f;

#define L_LEN 512
#define KW    5

// ---------------------------------------------------------------------------
// Gather: x0[b,l,c] = emb[features[b,l], c]
// ---------------------------------------------------------------------------
__global__ void ccnn_gather(const int* __restrict__ feats,
                            const float* __restrict__ emb,
                            float* __restrict__ x0, int n)
{
    int idx = blockIdx.x * blockDim.x + threadIdx.x;   // over B*L
    if (idx < n) {
        int t = feats[idx];
        #pragma unroll
        for (int c = 0; c < 8; ++c) x0[idx * 8 + c] = emb[t * 8 + c];
    }
}

// ---------------------------------------------------------------------------
// One CCNN filter pass: x_out[b,i,d] = sum_{j=i-5..i-1, c} x_in[b,i,c]*KV(dt)[c,d]
// One wave (32 threads) per (b, 16-row tile of i). Safe in-place (xin==xout):
// rows are block-exclusive and staged through LDS before any store.
// ---------------------------------------------------------------------------
__global__ __launch_bounds__(32) void ccnn_filter(
    const float* __restrict__ times,   // (B, L)
    const float* __restrict__ xin,     // (B, L, 8)
    const float* __restrict__ W1, const float* __restrict__ B1,  // (16),(16)
    const float* __restrict__ W2, const float* __restrict__ B2,  // (16,32),(32)
    const float* __restrict__ W3, const float* __restrict__ B3,  // (32,16),(16)
    const float* __restrict__ W4, const float* __restrict__ B4,  // (16,64),(64)
    float* __restrict__ xout)          // (B, L, 8)
{
    __shared__ float ts[32];           // times window [i0-5 .. i0+15], clamped
    __shared__ float xs[16 * 8];       // x tile
    __shared__ float h2s[16 * 32];     // layout bounce: D-layout -> A-layout
    __shared__ float h3s[16 * 16];

    const int lane = threadIdx.x;      // 0..31
    const int b    = blockIdx.x >> 5;  // L/16 = 32 tiles per batch
    const int tile = blockIdx.x & 31;
    const int i0   = tile * 16;
    const int p    = lane & 15;        // row (M) / column (N) index within 16
    const int half = lane >> 4;        // lane half selects K-range / row block

    // ---- stage times window + x tile into LDS ----
    if (lane < 16 + KW) {
        int gi = i0 - KW + lane;
        gi = gi < 0 ? 0 : gi;
        ts[lane] = times[b * L_LEN + gi];
    }
    for (int q = lane; q < 16 * 8; q += 32)
        xs[q] = xin[(b * L_LEN + i0) * 8 + q];
    __syncthreads();

    // ---- per-lane constant fragments (built once, reused for all 5 deltas) ----
    // W1/B1 for features m = half*8 + 0..7 (feeds the A-layout H1 build)
    float w1v[8], b1v[8];
    #pragma unroll
    for (int j = 0; j < 8; ++j) { w1v[j] = W1[half * 8 + j]; b1v[j] = B1[half * 8 + j]; }

    // B-fragment layout (f16 32x16): lane holds N = p, K = (v<4 ? half*8 + 2v : 16 + half*8 + 2(v-4)) + {0,1}
    // W2 is 16x32 -> two N tiles, K real 0..15 (upper half zero-padded)
    v16h w2b[2];
    #pragma unroll
    for (int t = 0; t < 2; ++t)
        #pragma unroll
        for (int v = 0; v < 8; ++v) {
            int k0 = (v < 4) ? half * 8 + 2 * v : 16 + half * 8 + 2 * (v - 4);
            float m = (k0 < 16) ? 1.f : 0.f;
            int kc0 = k0 < 15 ? k0 : 15, kc1 = (k0 + 1) < 15 ? (k0 + 1) : 15;
            w2b[t][2 * v]     = (_Float16)(m * W2[kc0 * 32 + t * 16 + p]);
            w2b[t][2 * v + 1] = (_Float16)(m * W2[kc1 * 32 + t * 16 + p]);
        }
    // W3 is 32x16 -> full K=32, one tile
    v16h w3b;
    #pragma unroll
    for (int v = 0; v < 8; ++v) {
        int k0 = (v < 4) ? half * 8 + 2 * v : 16 + half * 8 + 2 * (v - 4);
        w3b[2 * v]     = (_Float16)W3[k0 * 16 + p];
        w3b[2 * v + 1] = (_Float16)W3[(k0 + 1) * 16 + p];
    }
    // W4 is 16x64 -> four N tiles, K real 0..15
    v16h w4b[4];
    #pragma unroll
    for (int t = 0; t < 4; ++t)
        #pragma unroll
        for (int v = 0; v < 8; ++v) {
            int k0 = (v < 4) ? half * 8 + 2 * v : 16 + half * 8 + 2 * (v - 4);
            float m = (k0 < 16) ? 1.f : 0.f;
            int kc0 = k0 < 15 ? k0 : 15, kc1 = (k0 + 1) < 15 ? (k0 + 1) : 15;
            w4b[t][2 * v]     = (_Float16)(m * W4[kc0 * 64 + t * 16 + p]);
            w4b[t][2 * v + 1] = (_Float16)(m * W4[kc1 * 64 + t * 16 + p]);
        }

    // biases broadcast in C/D layout (column N = tile*16 + p, same for all rows)
    float bias2[2] = { B2[p], B2[16 + p] };
    float bias3    = B3[p];
    float bias4[4] = { B4[p], B4[16 + p], B4[32 + p], B4[48 + p] };

    // x values for the final contraction: this lane's (row r+8*half, c = 2t+(p>>3))
    float xreg[8][4];
    #pragma unroll
    for (int r = 0; r < 8; ++r)
        #pragma unroll
        for (int t = 0; t < 4; ++t)
            xreg[r][t] = xs[(r + 8 * half) * 8 + 2 * t + (p >> 3)];

    float acc[8];
    #pragma unroll
    for (int r = 0; r < 8; ++r) acc[r] = 0.f;

    // ---- loop over the 5 band offsets j = i - delta ----
    for (int delta = 1; delta <= KW; ++delta) {
        // dt for this lane's row p
        float s = fmaxf(ts[p + KW] - ts[p + KW - delta], 0.f);

        // H1 in A-fragment layout: M = p, K(feature) = half*8 + 2v + {0,1}; K>=16 zero
        v16h a1;
        #pragma unroll
        for (int v = 0; v < 4; ++v) {
            a1[2 * v]     = (_Float16)fmaxf(s * w1v[2 * v]     + b1v[2 * v],     0.f);
            a1[2 * v + 1] = (_Float16)fmaxf(s * w1v[2 * v + 1] + b1v[2 * v + 1], 0.f);
            a1[8 + 2 * v]     = (_Float16)0.f;
            a1[8 + 2 * v + 1] = (_Float16)0.f;
        }

        // H2 = relu(H1 @ W2 + B2)  (two 16-col tiles), bounce D->A via LDS
        __syncthreads();
        #pragma unroll
        for (int t = 0; t < 2; ++t) {
            v8f c;
            #pragma unroll
            for (int r = 0; r < 8; ++r) c[r] = bias2[t];
            c = __builtin_amdgcn_wmma_f32_16x16x32_f16(false, a1, false, w2b[t],
                                                       (short)0, c, false, false);
            #pragma unroll
            for (int r = 0; r < 8; ++r)
                h2s[(r + 8 * half) * 32 + t * 16 + p] = fmaxf(c[r], 0.f);
        }
        __syncthreads();

        // rebuild H2 as A fragment (K = 0..31 real)
        v16h a2;
        #pragma unroll
        for (int v = 0; v < 4; ++v) {
            int k  = half * 8 + 2 * v;
            int k2 = 16 + half * 8 + 2 * v;
            a2[2 * v]         = (_Float16)h2s[p * 32 + k];
            a2[2 * v + 1]     = (_Float16)h2s[p * 32 + k + 1];
            a2[8 + 2 * v]     = (_Float16)h2s[p * 32 + k2];
            a2[8 + 2 * v + 1] = (_Float16)h2s[p * 32 + k2 + 1];
        }

        // H3 = relu(H2 @ W3 + B3)  — one full-K WMMA
        {
            v8f c;
            #pragma unroll
            for (int r = 0; r < 8; ++r) c[r] = bias3;
            c = __builtin_amdgcn_wmma_f32_16x16x32_f16(false, a2, false, w3b,
                                                       (short)0, c, false, false);
            __syncthreads();
            #pragma unroll
            for (int r = 0; r < 8; ++r)
                h3s[(r + 8 * half) * 16 + p] = fmaxf(c[r], 0.f);
            __syncthreads();
        }

        // H3 as A fragment (K real 0..15, upper half zero)
        v16h a3;
        #pragma unroll
        for (int v = 0; v < 4; ++v) {
            int k = half * 8 + 2 * v;
            a3[2 * v]         = (_Float16)h3s[p * 16 + k];
            a3[2 * v + 1]     = (_Float16)h3s[p * 16 + k + 1];
            a3[8 + 2 * v]     = (_Float16)0.f;
            a3[8 + 2 * v + 1] = (_Float16)0.f;
        }

        // KV tiles (4 WMMAs) + masked contraction with x
        #pragma unroll
        for (int t = 0; t < 4; ++t) {
            v8f c;
            #pragma unroll
            for (int r = 0; r < 8; ++r) c[r] = bias4[t];
            c = __builtin_amdgcn_wmma_f32_16x16x32_f16(false, a3, false, w4b[t],
                                                       (short)0, c, false, false);
            #pragma unroll
            for (int r = 0; r < 8; ++r) {
                int   row = r + 8 * half;                       // i-row within tile
                float m   = (i0 + row - delta >= 0) ? 1.f : 0.f; // band validity
                acc[r] += m * xreg[r][t] * c[r];
            }
        }
    }

    // combine even/odd c-halves: partner lane is lane ^ 8 (same row set, same dcol)
    #pragma unroll
    for (int r = 0; r < 8; ++r)
        acc[r] += __shfl_xor(acc[r], 8, 32);

    // lanes with p<8 (per half) write out[b, i0 + r + 8*half, p&7]
    if ((p & 8) == 0) {
        #pragma unroll
        for (int r = 0; r < 8; ++r)
            xout[(b * L_LEN + i0 + r + 8 * half) * 8 + (p & 7)] = acc[r];
    }
}

// ---------------------------------------------------------------------------
// Launch: gather -> filter0 (in place on d_out) -> filter1 (in place)
// ---------------------------------------------------------------------------
extern "C" void kernel_launch(void* const* d_in, const int* in_sizes, int n_in,
                              void* d_out, int out_size, void* d_ws, size_t ws_size,
                              hipStream_t stream)
{
    (void)in_sizes; (void)n_in; (void)out_size; (void)d_ws; (void)ws_size;

    const float* times = (const float*)d_in[0];
    const int*   feats = (const int*)  d_in[1];
    const float* emb   = (const float*)d_in[2];
    const float* W1    = (const float*)d_in[3];
    const float* B1    = (const float*)d_in[4];
    const float* W2    = (const float*)d_in[5];
    const float* B2    = (const float*)d_in[6];
    const float* W3    = (const float*)d_in[7];
    const float* B3    = (const float*)d_in[8];
    const float* W4    = (const float*)d_in[9];
    const float* B4    = (const float*)d_in[10];
    float* out = (float*)d_out;

    const int BxL = 4 * L_LEN;              // 2048
    ccnn_gather<<<(BxL + 255) / 256, 256, 0, stream>>>(feats, emb, out, BxL);

    const int nblk = 4 * (L_LEN / 16);      // 128 waves
    // filter 0 (per-filter weight strides: W1:16 B1:16 W2:512 B2:32 W3:512 B3:16 W4:1024 B4:64)
    ccnn_filter<<<nblk, 32, 0, stream>>>(times, out,
        W1, B1, W2, B2, W3, B3, W4, B4, out);
    // filter 1
    ccnn_filter<<<nblk, 32, 0, stream>>>(times, out,
        W1 + 16, B1 + 16, W2 + 512, B2 + 32, W3 + 512, B3 + 16, W4 + 1024, B4 + 64, out);
}
